// GeneralizedCrossMessage_81363860456156
// MI455X (gfx1250) — matrix operand order repaired and checked by reading, not compiled, not run on hardware
//
#include <hip/hip_runtime.h>
#include <hip/hip_bf16.h>

// Problem constants (match reference)
#define NB 4
#define NS 512
#define NH 1024
#define ND 24
#define NF 96      // 4*D pair features
#define NMH 96     // MLP hidden
#define NVIN 2048  // 2*H
#define NVH 1024   // value hidden
#define NEGV (-3.4028234663852886e38f)

typedef __attribute__((ext_vector_type(16))) __bf16 v16bf;
typedef __attribute__((ext_vector_type(8)))  __bf16 v8bf;
typedef __attribute__((ext_vector_type(8)))  float  v8f;

static __device__ __forceinline__ v8f vzero() {
    v8f z;
#pragma unroll
    for (int i = 0; i < 8; ++i) z[i] = 0.0f;
    return z;
}

// A-fragment (16x32 bf16, M rows over lanes 0..15): per ISA table, lanes 0-15
// hold K = 0..7 (elems 0-7) and K = 16..23 (elems 8-15); lanes 16-31 hold
// K = 8..15 and K = 24..31.  ptr points at row start + k_base.
static __device__ __forceinline__ v16bf load_a_frag(const __bf16* ptr, int lane) {
    int koff = (lane & 16) ? 8 : 0;
    union { v16bf v; v8bf h[2]; } u;
    u.h[0] = *(const v8bf*)(ptr + koff);
    u.h[1] = *(const v8bf*)(ptr + koff + 16);
    return u.v;
}

// B-fragment (32x16 bf16, N cols over lanes 0..15): lanes 0-15 hold K=0..7
// and K=16..23, lanes 16-31 hold K=8..15 and K=24..31.  ptr = col(row-of-W) + k_base.
static __device__ __forceinline__ v16bf load_b_frag(const __bf16* ptr, int lane) {
    int koff = (lane & 16) ? 8 : 0;
    union { v16bf v; v8bf h[2]; } u;
    u.h[0] = *(const v8bf*)(ptr + koff);
    u.h[1] = *(const v8bf*)(ptr + koff + 16);
    return u.v;
}

static __device__ __forceinline__ v8f wmma_bf16(v16bf a, v16bf b, v8f c) {
    return __builtin_amdgcn_wmma_f32_16x16x32_bf16(false, a, false, b, (short)0, c,
                                                   false, false);
}

// ---------------------------------------------------------------- prep kernels
__global__ void f32_to_bf16_kernel(const float* __restrict__ src,
                                   __bf16* __restrict__ dst, int n) {
    int i = blockIdx.x * blockDim.x + threadIdx.x;
    if (i < n) dst[i] = (__bf16)src[i];
}

__global__ void hi_transpose_kernel(const float* __restrict__ Hi,
                                    __bf16* __restrict__ HiT) {
    int i = blockIdx.x * blockDim.x + threadIdx.x;  // over NB*NS*NH
    if (i >= NB * NS * NH) return;
    int h = i % NH;
    int bs = i / NH;
    int s = bs % NS;
    int b = bs / NS;
    HiT[((size_t)b * NH + h) * NS + s] = (__bf16)Hi[i];
}

__global__ void hj_pack_kernel(const float* __restrict__ Hj,
                               __bf16* __restrict__ msg_in) {
    int i = blockIdx.x * blockDim.x + threadIdx.x;  // over NB*NS*NH
    if (i >= NB * NS * NH) return;
    int h = i % NH;
    int tok = i / NH;
    msg_in[(size_t)tok * NVIN + NH + h] = (__bf16)Hj[i];
}

// Zj[b,s,d] = sum_h Hj * Wpj[d,h]; Zi likewise.  One thread per output.
__global__ void proj_kernel(const float* __restrict__ Hj,
                            const float* __restrict__ Hi,
                            const float* __restrict__ Wpj,
                            const float* __restrict__ Wpi,
                            float* __restrict__ Zj, float* __restrict__ Zi) {
    int i = blockIdx.x * blockDim.x + threadIdx.x;
    int total = NB * NS * ND;
    if (i >= 2 * total) return;
    const float* Hsrc;
    const float* W;
    float* Z;
    int j = i;
    if (i < total) { Hsrc = Hj; W = Wpj; Z = Zj; }
    else           { j -= total; Hsrc = Hi; W = Wpi; Z = Zi; }
    int d = j % ND;
    int row = j / ND;
    const float* hrow = Hsrc + (size_t)row * NH;
    const float* wrow = W + (size_t)d * NH;
    float acc = 0.0f;
    for (int h = 0; h < NH; ++h) acc = fmaf(hrow[h], wrow[h], acc);
    Z[j] = acc;
}

// ------------------------------------------------------------- score + softmax
// One block per (b,s).  8 waves, each handles 4 tiles of 16 t-values:
// features built in LDS, pair-MLP via WMMA (M=16 t, N=96 hidden, K=96),
// ReLU + W2 dot via shuffle reduce, then block-wide masked softmax -> bf16 probs.
__global__ void score_kernel(const float* __restrict__ Zj,
                             const float* __restrict__ Zi,
                             const __bf16* __restrict__ W1bf,
                             const float* __restrict__ b1,
                             const float* __restrict__ W2,
                             const float* __restrict__ b2,
                             const float* __restrict__ attn_mask,
                             const unsigned char* __restrict__ pair_mask,
                             __bf16* __restrict__ probs) {
    __shared__ float s_zj[ND];
    __shared__ float s_b1[NMH];
    __shared__ float s_w2[NMH];
    __shared__ __bf16 s_feat[8][16][NF];
    __shared__ float s_logits[NS];
    __shared__ float s_red[256];

    int tid = threadIdx.x;
    int s = blockIdx.x % NS;
    int b = blockIdx.x / NS;

    if (tid < ND) s_zj[tid] = Zj[((size_t)b * NS + s) * ND + tid];
    if (tid < NMH) { s_b1[tid] = b1[tid]; s_w2[tid] = W2[tid]; }
    __syncthreads();

    int wave = tid >> 5;
    int lane = tid & 31;
    float b2v = b2[0];

    for (int it = 0; it < 4; ++it) {
        int tbase = (wave * 4 + it) * 16;
        __bf16* feat = &s_feat[wave][0][0];
        // Build 16 x 96 bf16 feature tile: [Zj, Zi, Zj*Zi, |Zj-Zi|]
        for (int idx = lane; idx < 16 * ND; idx += 32) {
            int tl = idx / ND;
            int d = idx - tl * ND;
            float zj = s_zj[d];
            float zi = Zi[((size_t)b * NS + tbase + tl) * ND + d];
            __bf16* fr = feat + tl * NF;
            fr[d]          = (__bf16)zj;
            fr[ND + d]     = (__bf16)zi;
            fr[2 * ND + d] = (__bf16)(zj * zi);
            fr[3 * ND + d] = (__bf16)fabsf(zj - zi);
        }
        asm volatile("s_wait_dscnt 0" ::: "memory");   // CDNA5 split LDS counter
        __builtin_amdgcn_wave_barrier();

        v8f acc[6];
#pragma unroll
        for (int n = 0; n < 6; ++n) acc[n] = vzero();
#pragma unroll
        for (int ks = 0; ks < 3; ++ks) {
            v16bf a = load_a_frag(feat + (lane & 15) * NF + ks * 32, lane);
#pragma unroll
            for (int n = 0; n < 6; ++n) {
                v16bf bb = load_b_frag(
                    W1bf + (size_t)(16 * n + (lane & 15)) * NF + ks * 32, lane);
                acc[n] = wmma_bf16(a, bb, acc[n]);
            }
        }

        int col = lane & 15;
        int hs = lane >> 4;
#pragma unroll
        for (int r = 0; r < 8; ++r) {
            float p = 0.0f;
#pragma unroll
            for (int n = 0; n < 6; ++n) {
                float hval = acc[n][r] + s_b1[16 * n + col];
                hval = fmaxf(hval, 0.0f);
                p += hval * s_w2[16 * n + col];
            }
#pragma unroll
            for (int off = 1; off < 16; off <<= 1) p += __shfl_xor(p, off, 32);
            if (col == 0) {
                int t = tbase + r + 8 * hs;
                float lg = p + b2v;
                lg += (1.0f - attn_mask[(size_t)b * NS + t]) * NEGV;
                if (!pair_mask[((size_t)b * NS + s) * NS + t]) lg = NEGV;
                s_logits[t] = lg;
            }
        }
        __builtin_amdgcn_wave_barrier();
    }
    __syncthreads();

    // block-wide softmax over 512 logits (each thread owns 2)
    float l0 = s_logits[tid];
    float l1 = s_logits[tid + 256];
    s_red[tid] = fmaxf(l0, l1);
    __syncthreads();
    for (int off = 128; off > 0; off >>= 1) {
        if (tid < off) s_red[tid] = fmaxf(s_red[tid], s_red[tid + off]);
        __syncthreads();
    }
    float mx = s_red[0];
    __syncthreads();
    float e0 = expf(l0 - mx);
    float e1 = expf(l1 - mx);
    s_red[tid] = e0 + e1;
    __syncthreads();
    for (int off = 128; off > 0; off >>= 1) {
        if (tid < off) s_red[tid] += s_red[tid + off];
        __syncthreads();
    }
    float inv = 1.0f / s_red[0];
    size_t pb = ((size_t)b * NS + s) * NS;
    probs[pb + tid] = (__bf16)(e0 * inv);
    probs[pb + tid + 256] = (__bf16)(e1 * inv);
}

// --------------------------------------------------------- ctx = probs @ H_i
// Block: 16 s-rows x 128 h-cols; 8 waves, one 16x16 tile each; K = 512 (t).
__global__ void ctx_kernel(const __bf16* __restrict__ probs,
                           const __bf16* __restrict__ HiT,
                           __bf16* __restrict__ msg_in) {
    int wave = threadIdx.x >> 5;
    int lane = threadIdx.x & 31;
    int bid = blockIdx.x;
    int hblk = bid & 7;
    int rest = bid >> 3;
    int stile = rest & 31;
    int b = rest >> 5;
    int sbase = stile * 16;
    int nbase = hblk * 128 + wave * 16;

    const __bf16* arow = probs + ((size_t)b * NS + sbase + (lane & 15)) * NS;
    const __bf16* brow = HiT + ((size_t)b * NH + nbase + (lane & 15)) * NS;

    v8f acc = vzero();
    for (int ks = 0; ks < 16; ++ks) {
        v16bf a = load_a_frag(arow + ks * 32, lane);
        v16bf bb = load_b_frag(brow + ks * 32, lane);
        acc = wmma_bf16(a, bb, acc);
    }
    int col = nbase + (lane & 15);
#pragma unroll
    for (int r = 0; r < 8; ++r) {
        int m = r + 8 * (lane >> 4);
        msg_in[(size_t)(b * NS + sbase + m) * NVIN + col] = (__bf16)acc[r];
    }
}

// ------------------------------------------- hv = relu(msg_in @ Wv1^T + bv1)
__global__ void value1_kernel(const __bf16* __restrict__ msg_in,
                              const __bf16* __restrict__ Wv1bf,
                              const float* __restrict__ bv1,
                              __bf16* __restrict__ hv) {
    int wave = threadIdx.x >> 5;
    int lane = threadIdx.x & 31;
    int nblk = blockIdx.x & 7;
    int mtile = blockIdx.x >> 3;
    int mbase = mtile * 16;
    int nbase = nblk * 128 + wave * 16;

    const __bf16* arow = msg_in + (size_t)(mbase + (lane & 15)) * NVIN;
    const __bf16* brow = Wv1bf + (size_t)(nbase + (lane & 15)) * NVIN;

    v8f acc = vzero();
    for (int ks = 0; ks < 64; ++ks) {
        if (ks + 8 < 64) __builtin_prefetch(arow + (ks + 8) * 32, 0, 0);
        v16bf a = load_a_frag(arow + ks * 32, lane);
        v16bf bb = load_b_frag(brow + ks * 32, lane);
        acc = wmma_bf16(a, bb, acc);
    }
    int col = nbase + (lane & 15);
    float bias = bv1[col];
#pragma unroll
    for (int r = 0; r < 8; ++r) {
        int m = r + 8 * (lane >> 4);
        hv[(size_t)(mbase + m) * NVH + col] = (__bf16)fmaxf(acc[r] + bias, 0.0f);
    }
}

// ------------------------------------- out = alpha * (hv @ Wv2^T + bv2), f32
__global__ void value2_kernel(const __bf16* __restrict__ hv,
                              const __bf16* __restrict__ Wv2bf,
                              const float* __restrict__ bv2,
                              const float* __restrict__ alpha,
                              float* __restrict__ out) {
    int wave = threadIdx.x >> 5;
    int lane = threadIdx.x & 31;
    int nblk = blockIdx.x & 7;
    int mtile = blockIdx.x >> 3;
    int mbase = mtile * 16;
    int nbase = nblk * 128 + wave * 16;

    const __bf16* arow = hv + (size_t)(mbase + (lane & 15)) * NVH;
    const __bf16* brow = Wv2bf + (size_t)(nbase + (lane & 15)) * NVH;

    v8f acc = vzero();
    for (int ks = 0; ks < 32; ++ks) {
        v16bf a = load_a_frag(arow + ks * 32, lane);
        v16bf bb = load_b_frag(brow + ks * 32, lane);
        acc = wmma_bf16(a, bb, acc);
    }
    int col = nbase + (lane & 15);
    float bias = bv2[col];
    float al = alpha[0];
#pragma unroll
    for (int r = 0; r < 8; ++r) {
        int m = r + 8 * (lane >> 4);
        out[(size_t)(mbase + m) * NH + col] = al * (acc[r] + bias);
    }
}

// ------------------------------------------------------------------- launcher
extern "C" void kernel_launch(void* const* d_in, const int* in_sizes, int n_in,
                              void* d_out, int out_size, void* d_ws, size_t ws_size,
                              hipStream_t stream) {
    const float* Hj = (const float*)d_in[0];
    const float* Hi = (const float*)d_in[1];
    const float* attn = (const float*)d_in[2];
    const unsigned char* pmask = (const unsigned char*)d_in[3];
    const float* Wpj = (const float*)d_in[4];
    const float* Wpi = (const float*)d_in[5];
    const float* W1 = (const float*)d_in[6];
    const float* b1 = (const float*)d_in[7];
    const float* W2 = (const float*)d_in[8];
    const float* b2 = (const float*)d_in[9];
    const float* Wv1 = (const float*)d_in[10];
    const float* bv1 = (const float*)d_in[11];
    const float* Wv2 = (const float*)d_in[12];
    const float* bv2 = (const float*)d_in[13];
    const float* alpha = (const float*)d_in[14];

    // workspace layout (bytes, 256-aligned); total ~24.4 MB, L2-resident
    char* ws = (char*)d_ws;
    float*  Zj     = (float*)(ws + 0);          //  196608 B
    float*  Zi     = (float*)(ws + 196608);     //  196608 B
    __bf16* W1bf   = (__bf16*)(ws + 393216);    //   18432 B
    __bf16* probs  = (__bf16*)(ws + 411648);    // 2097152 B
    __bf16* HiT    = (__bf16*)(ws + 2508800);   // 4194304 B  [b][h][t]
    __bf16* msg_in = (__bf16*)(ws + 6703104);   // 8388608 B  [tok][ctx|Hj]
    __bf16* Wv1bf  = (__bf16*)(ws + 15091712);  // 4194304 B
    __bf16* Wv2bf  = (__bf16*)(ws + 19286016);  // 2097152 B
    __bf16* hv     = (__bf16*)(ws + 21383168);  // 4194304 B

    f32_to_bf16_kernel<<<(NMH * NF + 255) / 256, 256, 0, stream>>>(W1, W1bf, NMH * NF);
    f32_to_bf16_kernel<<<(NVH * NVIN + 255) / 256, 256, 0, stream>>>(Wv1, Wv1bf, NVH * NVIN);
    f32_to_bf16_kernel<<<(NH * NVH + 255) / 256, 256, 0, stream>>>(Wv2, Wv2bf, NH * NVH);
    hi_transpose_kernel<<<(NB * NS * NH) / 256, 256, 0, stream>>>(Hi, HiT);
    hj_pack_kernel<<<(NB * NS * NH) / 256, 256, 0, stream>>>(Hj, msg_in);
    proj_kernel<<<(2 * NB * NS * ND + 255) / 256, 256, 0, stream>>>(Hj, Hi, Wpj, Wpi, Zj, Zi);

    score_kernel<<<NB * NS, 256, 0, stream>>>(Zj, Zi, W1bf, b1, W2, b2, attn, pmask, probs);
    ctx_kernel<<<NB * (NS / 16) * (NH / 128), 256, 0, stream>>>(probs, HiT, msg_in);
    value1_kernel<<<(NB * NS / 16) * (NVH / 128), 256, 0, stream>>>(msg_in, Wv1bf, bv1, hv);
    value2_kernel<<<(NB * NS / 16) * (NH / 128), 256, 0, stream>>>(hv, Wv2bf, bv2, alpha, (float*)d_out);
}
